// SBERTaModel_1443109011852
// MI455X (gfx1250) — compile-verified, matrix operand order checked by reading, not run on hardware
//
#include <hip/hip_runtime.h>
#include <hip/hip_bf16.h>
#include <cstdint>

// ---------------------------------------------------------------------------
// SBERTa forward for MI455X (gfx1250): WMMA f16 GEMMs + TDM async tile loads.
// All GEMM B-operands are stored K-contiguous (transposed) so every WMMA
// fragment is built from two ds_load_b128 per lane.
// ---------------------------------------------------------------------------

typedef __attribute__((ext_vector_type(16))) _Float16 v16h;
typedef __attribute__((ext_vector_type(8)))  _Float16 v8h;
typedef __attribute__((ext_vector_type(8)))  float    v8f;

#if defined(__has_builtin)
#  if __has_builtin(__builtin_amdgcn_tensor_load_to_lds) && __has_builtin(__builtin_amdgcn_s_wait_tensorcnt)
#    define USE_TDM 1
#  endif
#endif
#ifndef USE_TDM
#  define USE_TDM 0
#endif

// Model constants (fixed problem size)
#define Bc   8
#define Tc   512
#define Dc   768
#define NHc  12
#define DHc  64
#define FFc  3072
#define K16  16
#define Mc   (Bc * Tc)        // 4096
#define BHc  (Bc * NHc)       // 96

// ---------------------------------------------------------------------------
// TDM descriptor helpers (D# group0/group1 per CDNA5 ISA ch.8)
// ---------------------------------------------------------------------------
#if USE_TDM
typedef __attribute__((ext_vector_type(4))) unsigned int tdm_u32x4;
typedef __attribute__((ext_vector_type(8))) int          tdm_i32x8;
typedef __attribute__((ext_vector_type(4))) int          tdm_i32x4;

__device__ __forceinline__ unsigned lds_byteoff(const void* p) {
  // low 32 bits of a generic (LDS-aperture) address are the LDS byte address
  return (unsigned)(uintptr_t)p;
}

// 2D tile load: tile_w contiguous f16 elements per row, tile_h rows,
// row pitch = row_stride_elems (f16 units). Tensor dims == tile dims (no OOB).
__device__ __forceinline__ void tdm_load_2d(const void* gsrc, unsigned lds_addr,
                                            unsigned tile_w, unsigned tile_h,
                                            unsigned long long row_stride_elems)
{
  unsigned long long ga = (unsigned long long)(uintptr_t)gsrc;
  tdm_u32x4 g0;
  g0[0] = 1u;                                              // count=1, user mode
  g0[1] = lds_addr;                                        // lds_addr [63:32]
  g0[2] = (unsigned)ga;                                    // global_addr lo
  g0[3] = (unsigned)((ga >> 32) & 0x1ffffffull) | (2u << 30); // addr hi | type=2
  tdm_i32x8 g1;
  g1[0] = (int)(1u << 16);                                 // data_size=1 (2 bytes)
  g1[1] = (int)((tile_w & 0xffffu) << 16);                 // tensor_dim0 lo16
  g1[2] = (int)((tile_w >> 16) | ((tile_h & 0xffffu) << 16)); // dim0 hi | dim1 lo
  g1[3] = (int)((tile_h >> 16) | (tile_w << 16));          // dim1 hi | tile_dim0
  g1[4] = (int)(tile_h & 0xffffu);                         // tile_dim1 ; tile_dim2=0
  g1[5] = (int)(row_stride_elems & 0xffffffffull);         // dim0_stride lo32
  g1[6] = (int)((row_stride_elems >> 32) & 0xffffull);     // dim0_stride hi16
  g1[7] = 0;
  tdm_i32x4 z4 = {0, 0, 0, 0};
  tdm_i32x8 z8 = {0, 0, 0, 0, 0, 0, 0, 0};
  // clang-23 / therock toolchain: 6-arg form (extra int32x8 before cpol)
  __builtin_amdgcn_tensor_load_to_lds(g0, g1, z4, z4, z8, 0);
}
#endif

// ---------------------------------------------------------------------------
// WMMA fragment builder (16x16x32 f16, layout per ISA 7.12.2).
// Lane data = two contiguous 16-byte chunks -> two ds_load_b128.
// rowbase points at element [row][0] of a row-major (K-contiguous) tile.
// ---------------------------------------------------------------------------
__device__ __forceinline__ v16h frag_rowmajor(const _Float16* rowbase, int kk0, int lane) {
  const int half = (lane >> 4) & 1;
  const _Float16* r = rowbase + kk0 + half * 8;
  const v8h lo = *(const v8h*)(r);        // K = base+0..7
  const v8h hi = *(const v8h*)(r + 16);   // K = base+16..23
  v16h f;
#pragma unroll
  for (int i = 0; i < 8; ++i) { f[i] = lo[i]; f[8 + i] = hi[i]; }
  return f;
}

// ---------------------------------------------------------------------------
// Kernels
// ---------------------------------------------------------------------------
__global__ void __launch_bounds__(256)
f32_to_f16_kernel(const float* __restrict__ src, _Float16* __restrict__ dst, long n) {
  long i = (long)blockIdx.x * blockDim.x + threadIdx.x;
  long stride = (long)gridDim.x * blockDim.x;
  for (; i < n; i += stride) dst[i] = (_Float16)src[i];
}

// dst[l][n][k] = (f16) src[l][k][n]  (weight transpose during convert)
__global__ void __launch_bounds__(256)
f32_to_f16_T_kernel(const float* __restrict__ src, _Float16* __restrict__ dst,
                    int Kd, int Nd, int nl) {
  long tot = (long)nl * Kd * Nd;
  long stride = (long)gridDim.x * blockDim.x;
  for (long idx = (long)blockIdx.x * blockDim.x + threadIdx.x; idx < tot; idx += stride) {
    long l = idx / ((long)Kd * Nd);
    long rem = idx - l * (long)Kd * Nd;
    int n = (int)(rem / Kd);
    int k = (int)(rem - (long)n * Kd);
    dst[idx] = (_Float16)src[(l * Kd + k) * (long)Nd + n];
  }
}

// h[row,:] = LN(tok_emb[id] + pos_emb[t]) * g + b  (wave per row)
__global__ void __launch_bounds__(256)
embed_ln_kernel(const int* __restrict__ ids, const float* __restrict__ tok,
                const float* __restrict__ pos, const float* __restrict__ g,
                const float* __restrict__ b, float* __restrict__ H, int nrows) {
  int wid = blockIdx.x * (blockDim.x >> 5) + (threadIdx.x >> 5);
  if (wid >= nrows) return;
  int lane = threadIdx.x & 31;
  int t = wid % Tc;
  const float* tr = tok + (long)ids[wid] * Dc;
  const float* pr = pos + (long)t * Dc;
  float s = 0.f, s2 = 0.f;
  for (int c = lane; c < Dc; c += 32) { float v = tr[c] + pr[c]; s += v; s2 += v * v; }
  for (int off = 16; off; off >>= 1) { s += __shfl_xor(s, off); s2 += __shfl_xor(s2, off); }
  float m = s / Dc, var = s2 / Dc - m * m;
  float rs = rsqrtf(var + 1e-12f);
  float* hr = H + (long)wid * Dc;
  for (int c = lane; c < Dc; c += 32) {
    float v = tr[c] + pr[c];
    hr[c] = (v - m) * rs * g[c] + b[c];
  }
}

// LayerNorm rows (wave per row); outH (f16) / outF (f32) optional
__global__ void __launch_bounds__(256)
ln_rows_kernel(const float* __restrict__ X, const float* __restrict__ g,
               const float* __restrict__ b, _Float16* __restrict__ outH,
               float* __restrict__ outF, int nrows) {
  int wid = blockIdx.x * (blockDim.x >> 5) + (threadIdx.x >> 5);
  if (wid >= nrows) return;
  int lane = threadIdx.x & 31;
  const float* x = X + (long)wid * Dc;
  float s = 0.f, s2 = 0.f;
  for (int c = lane; c < Dc; c += 32) { float v = x[c]; s += v; s2 += v * v; }
  for (int off = 16; off; off >>= 1) { s += __shfl_xor(s, off); s2 += __shfl_xor(s2, off); }
  float m = s / Dc, var = s2 / Dc - m * m;
  float rs = rsqrtf(var + 1e-12f);
  for (int c = lane; c < Dc; c += 32) {
    float y = (x[c] - m) * rs * g[c] + b[c];
    if (outH) outH[(long)wid * Dc + c] = (_Float16)y;
    if (outF) outF[(long)wid * Dc + c] = y;
  }
}

// Generic WMMA GEMM: C[M,N] = A[M,K] x Bt[N,K]^T (both f16, K-contiguous).
// Block tile 128x64, BK=32, 4 waves; each wave: 32 rows x 64 cols = 8 WMMA/step.
// flags: 1=+bias[col], 2=GELU(exact), 4=residual in-place into resid(f32),
//        8=(b,h) batch decomposition for B/C, 16=store outH transposed [B][Dc][Tc].
__global__ void __launch_bounds__(128)
gemm_f16_kernel(const _Float16* __restrict__ A, const _Float16* __restrict__ Bt,
                const float* __restrict__ bias, float* __restrict__ resid,
                _Float16* __restrict__ outH, float* __restrict__ outF,
                int M, int N, int K, int lda, int ldb, int ldc,
                long strideA, long strideB, long strideC,
                int nh, long dhB, long dhC, int flags) {
  __shared__ __align__(16) _Float16 sA[2][128 * 32];
  __shared__ __align__(16) _Float16 sB[2][64 * 32];
  const int lane = threadIdx.x & 31;
  const int wave = threadIdx.x >> 5;
  const int zb = blockIdx.z;
  long offA = (long)zb * strideA, offB, offC;
  if (flags & 8) {
    int b_ = zb / nh, h_ = zb % nh;
    offB = (long)b_ * strideB + (long)h_ * dhB;
    offC = (long)b_ * strideC + (long)h_ * dhC;
  } else {
    offB = (long)zb * strideB;
    offC = (long)zb * strideC;
  }
  const _Float16* Ab = A + offA + (long)blockIdx.y * 128 * lda;
  const _Float16* Bb = Bt + offB + (long)blockIdx.x * 64 * ldb;
  const int nk = K >> 5;
  v8f acc[2][4] = {};

#if USE_TDM
  if (wave == 0) {
    tdm_load_2d(Ab, lds_byteoff(&sA[0][0]), 32, 128, (unsigned long long)lda);
    tdm_load_2d(Bb, lds_byteoff(&sB[0][0]), 32, 64, (unsigned long long)ldb);
    __builtin_amdgcn_s_wait_tensorcnt(0);
  }
#else
  for (int t = threadIdx.x; t < 128 * 32; t += 128)
    sA[0][t] = Ab[(long)(t >> 5) * lda + (t & 31)];
  for (int t = threadIdx.x; t < 64 * 32; t += 128)
    sB[0][t] = Bb[(long)(t >> 5) * ldb + (t & 31)];
#endif
  __syncthreads();

  for (int kt = 0; kt < nk; ++kt) {
    const int cur = kt & 1, nxt = cur ^ 1;
#if USE_TDM
    if (wave == 0 && kt + 1 < nk) {
      tdm_load_2d(Ab + (long)(kt + 1) * 32, lds_byteoff(&sA[nxt][0]), 32, 128,
                  (unsigned long long)lda);
      tdm_load_2d(Bb + (long)(kt + 1) * 32, lds_byteoff(&sB[nxt][0]), 32, 64,
                  (unsigned long long)ldb);
    }
#else
    if (kt + 1 < nk) {
      for (int t = threadIdx.x; t < 128 * 32; t += 128)
        sA[nxt][t] = Ab[(long)(t >> 5) * lda + (kt + 1) * 32 + (t & 31)];
      for (int t = threadIdx.x; t < 64 * 32; t += 128)
        sB[nxt][t] = Bb[(long)(t >> 5) * ldb + (kt + 1) * 32 + (t & 31)];
    }
#endif
    v16h af[2];
#pragma unroll
    for (int mt = 0; mt < 2; ++mt)
      af[mt] = frag_rowmajor(&sA[cur][(wave * 32 + mt * 16 + (lane & 15)) * 32], 0, lane);
#pragma unroll
    for (int nt = 0; nt < 4; ++nt) {
      v16h bf = frag_rowmajor(&sB[cur][(nt * 16 + (lane & 15)) * 32], 0, lane);
#pragma unroll
      for (int mt = 0; mt < 2; ++mt)
        acc[mt][nt] = __builtin_amdgcn_wmma_f32_16x16x32_f16(false, af[mt], false, bf,
                                                             (short)0, acc[mt][nt], false, false);
    }
#if USE_TDM
    if (wave == 0) __builtin_amdgcn_s_wait_tensorcnt(0);
#endif
    __syncthreads();
  }

  // epilogue: C/D layout -> row = base + 8*half + r ; col = base + (lane&15)
  const int half = lane >> 4;
  const int col0 = blockIdx.x * 64 + (lane & 15);
#pragma unroll
  for (int mt = 0; mt < 2; ++mt) {
    const int row0 = blockIdx.y * 128 + wave * 32 + mt * 16 + half * 8;
#pragma unroll
    for (int nt = 0; nt < 4; ++nt) {
      const int col = col0 + nt * 16;
#pragma unroll
      for (int r = 0; r < 8; ++r) {
        const int row = row0 + r;
        float v = acc[mt][nt][r];
        if (flags & 1) v += bias[col];
        if (flags & 2) v = 0.5f * v * (1.0f + erff(v * 0.70710678118654752f));
        long ci = offC + (long)row * ldc + col;
        if (flags & 4) { v += resid[ci]; resid[ci] = v; }
        if (outH) {
          if (flags & 16) {
            // store transposed per batch: out[b][col][t], b=row/Tc, t=row%Tc
            long ti = (((long)(row >> 9)) * Dc + col) * Tc + (row & (Tc - 1));
            outH[ti] = (_Float16)v;
          } else {
            outH[ci] = (_Float16)v;
          }
        }
        if (outF) outF[ci] = v;
      }
    }
  }
}

// Attention scores: S[bh,i,j] = q_i.k_j/8 (+ phase2 biases) + mask additive.
// grid = (T/64 i-strips, NH, B), 128 threads. K chunks double-buffered via TDM.
__global__ void __launch_bounds__(128)
attn_scores_kernel(const _Float16* __restrict__ Q, const _Float16* __restrict__ Km,
                   float* __restrict__ S, const float* __restrict__ pc,
                   const float* __restrict__ p, const float* __restrict__ sw,
                   const float* __restrict__ gamma_arr, int layer,
                   const int* __restrict__ mask, int phase2) {
  __shared__ __align__(16) _Float16 sQ[64 * 64];
  __shared__ __align__(16) _Float16 sK[2][64 * 64];
  const int lane = threadIdx.x & 31, wave = threadIdx.x >> 5;
  const int b = blockIdx.z, h = blockIdx.y, bh = b * NHc + h;
  const int i0 = blockIdx.x * 64;
  const _Float16* Qb = Q + ((long)(b * Tc + i0)) * Dc + h * DHc;
  const _Float16* Kb = Km + ((long)b * Tc) * Dc + h * DHc;
#if USE_TDM
  if (wave == 0) {
    tdm_load_2d(Qb, lds_byteoff(sQ), 64, 64, (unsigned long long)Dc);
    tdm_load_2d(Kb, lds_byteoff(&sK[0][0]), 64, 64, (unsigned long long)Dc);
    __builtin_amdgcn_s_wait_tensorcnt(0);
  }
#else
  for (int t = threadIdx.x; t < 64 * 64; t += 128) {
    sQ[t]    = Qb[(long)(t >> 6) * Dc + (t & 63)];
    sK[0][t] = Kb[(long)(t >> 6) * Dc + (t & 63)];
  }
#endif
  __syncthreads();
  const _Float16* qrow = &sQ[(wave * 16 + (lane & 15)) * 64];
  v16h a0 = frag_rowmajor(qrow, 0, lane);
  v16h a1 = frag_rowmajor(qrow, 32, lane);
  const float gamma = phase2 ? gamma_arr[layer] : 0.0f;
  const int half = lane >> 4;
  for (int ch = 0; ch < Tc / 64; ++ch) {
    const int cur = ch & 1, nxt = cur ^ 1;
#if USE_TDM
    if (wave == 0 && ch + 1 < Tc / 64)
      tdm_load_2d(Kb + (long)(ch + 1) * 64 * Dc, lds_byteoff(&sK[nxt][0]), 64, 64,
                  (unsigned long long)Dc);
#else
    if (ch + 1 < Tc / 64)
      for (int t = threadIdx.x; t < 64 * 64; t += 128)
        sK[nxt][t] = Kb[((long)(ch + 1) * 64 + (t >> 6)) * Dc + (t & 63)];
#endif
#pragma unroll
    for (int jt = 0; jt < 4; ++jt) {
      const _Float16* krow = &sK[cur][(jt * 16 + (lane & 15)) * 64];
      v16h b0 = frag_rowmajor(krow, 0, lane);   // B^T columns == K rows
      v16h b1 = frag_rowmajor(krow, 32, lane);
      v8f acc = {};
      acc = __builtin_amdgcn_wmma_f32_16x16x32_f16(false, a0, false, b0, (short)0, acc, false, false);
      acc = __builtin_amdgcn_wmma_f32_16x16x32_f16(false, a1, false, b1, (short)0, acc, false, false);
      const int j = ch * 64 + jt * 16 + (lane & 15);
      float jbias = (1.0f - (float)mask[b * Tc + j]) * -10000.0f;
      const float* prow = p + ((long)(b * Tc + j)) * K16;
      if (phase2) jbias += gamma * sw[b * Tc + j];
#pragma unroll
      for (int r = 0; r < 8; ++r) {
        const int i = i0 + wave * 16 + half * 8 + r;
        float v = acc[r] * 0.125f;  // 1/sqrt(DH)
        if (phase2) {
          const float* pcr = pc + ((long)(bh * Tc + i)) * K16;
          float d = 0.f;
#pragma unroll
          for (int k = 0; k < K16; ++k) d += pcr[k] * prow[k];
          v += d;
        }
        v += jbias;
        S[((long)bh * Tc + i) * Tc + j] = v;
      }
    }
#if USE_TDM
    if (wave == 0) __builtin_amdgcn_s_wait_tensorcnt(0);
#endif
    __syncthreads();
  }
}

// Row softmax over T=512, f32 in -> f16 probs out (wave per row)
__global__ void __launch_bounds__(256)
softmax_rows_kernel(const float* __restrict__ S, _Float16* __restrict__ P, long nrows) {
  long wid = (long)blockIdx.x * (blockDim.x >> 5) + (threadIdx.x >> 5);
  if (wid >= nrows) return;
  int lane = threadIdx.x & 31;
  const float* r = S + wid * Tc;
  float ev[Tc / 32];
  float mx = -3.0e38f;
#pragma unroll
  for (int e = 0; e < Tc / 32; ++e) { ev[e] = r[lane + (e << 5)]; mx = fmaxf(mx, ev[e]); }
  for (int off = 16; off; off >>= 1) mx = fmaxf(mx, __shfl_xor(mx, off));
  float sum = 0.f;
#pragma unroll
  for (int e = 0; e < Tc / 32; ++e) { ev[e] = __expf(ev[e] - mx); sum += ev[e]; }
  for (int off = 16; off; off >>= 1) sum += __shfl_xor(sum, off);
  float inv = 1.f / sum;
  _Float16* pr = P + wid * Tc;
#pragma unroll
  for (int e = 0; e < Tc / 32; ++e) pr[lane + (e << 5)] = (_Float16)(ev[e] * inv);
}

// prototype row L2 normalize (one wave per k)
__global__ void __launch_bounds__(32)
proto_norm_kernel(const float* __restrict__ L, float* __restrict__ Ln) {
  int k = blockIdx.x, lane = threadIdx.x;
  float s = 0.f;
  for (int c = lane; c < Dc; c += 32) { float v = L[k * Dc + c]; s += v * v; }
  for (int off = 16; off; off >>= 1) s += __shfl_xor(s, off);
  float inv = rsqrtf(s);
  for (int c = lane; c < Dc; c += 32) Ln[k * Dc + c] = L[k * Dc + c] * inv;
}

// p[row,:] = softmax( LN(H_base)[row] @ Lnorm^T / tau )  (wave per row)
__global__ void __launch_bounds__(256)
proto_p_kernel(const _Float16* __restrict__ hn, const float* __restrict__ Ln,
               const float* __restrict__ logtau, float* __restrict__ Pout, int nrows) {
  int wid = blockIdx.x * (blockDim.x >> 5) + (threadIdx.x >> 5);
  if (wid >= nrows) return;
  int lane = threadIdx.x & 31;
  const _Float16* x = hn + (long)wid * Dc;
  float part[K16];
#pragma unroll
  for (int k = 0; k < K16; ++k) part[k] = 0.f;
  for (int c = lane; c < Dc; c += 32) {
    float xv = (float)x[c];
#pragma unroll
    for (int k = 0; k < K16; ++k) part[k] += xv * Ln[k * Dc + c];
  }
#pragma unroll
  for (int k = 0; k < K16; ++k)
    for (int off = 16; off; off >>= 1) part[k] += __shfl_xor(part[k], off);
  float tau = fmaxf(__expf(logtau[0]), 0.25f);
  float mx = -3.0e38f;
#pragma unroll
  for (int k = 0; k < K16; ++k) { part[k] /= tau; mx = fmaxf(mx, part[k]); }
  float sum = 0.f;
#pragma unroll
  for (int k = 0; k < K16; ++k) { part[k] = __expf(part[k] - mx); sum += part[k]; }
  float inv = 1.f / sum;
  if (lane == 0) {
#pragma unroll
    for (int k = 0; k < K16; ++k) Pout[(long)wid * K16 + k] = part[k] * inv;
  }
}

// s[b,t] = (t==0) ? 0 : 1 - p[b,t].p[b,t-1]
__global__ void __launch_bounds__(256)
switch_s_kernel(const float* __restrict__ P, float* __restrict__ S, int n) {
  int idx = blockIdx.x * blockDim.x + threadIdx.x;
  if (idx >= n) return;
  int t = idx % Tc;
  float v = 0.f;
  if (t > 0) {
    float d = 0.f;
#pragma unroll
    for (int k = 0; k < K16; ++k) d += P[(long)idx * K16 + k] * P[(long)(idx - 1) * K16 + k];
    v = 1.f - d;
  }
  S[idx] = v;
}

// pc[bh,i,k2] = sum_k p[b,i,k] * compat[layer,h,k,k2]
__global__ void __launch_bounds__(256)
pc_kernel(const float* __restrict__ P, const float* __restrict__ compat,
          float* __restrict__ pc, int layer) {
  int idx = blockIdx.x * blockDim.x + threadIdx.x;   // BH*T*16
  int k2 = idx & 15;
  int i  = (idx >> 4) & (Tc - 1);
  int bh = idx >> 13;
  if (bh >= BHc) return;
  int h = bh % NHc, b = bh / NHc;
  const float* pr = P + ((long)(b * Tc + i)) * K16;
  const float* C  = compat + (((long)layer * NHc + h) * K16) * K16 + k2;
  float acc = 0.f;
#pragma unroll
  for (int k = 0; k < K16; ++k) acc += pr[k] * C[k * K16];
  pc[(long)idx] = acc;
}

// ---------------------------------------------------------------------------
// Host launcher
// ---------------------------------------------------------------------------
extern "C" void kernel_launch(void* const* d_in, const int* in_sizes, int n_in,
                              void* d_out, int out_size, void* d_ws, size_t ws_size,
                              hipStream_t stream) {
  (void)in_sizes; (void)n_in; (void)out_size; (void)ws_size;
  // input order per setup_inputs() flattening
  const int*   ids   = (const int*)d_in[0];
  const int*   amask = (const int*)d_in[1];
  const float* tok   = (const float*)d_in[2];
  const float* pos   = (const float*)d_in[3];
  const float* elng  = (const float*)d_in[4];
  const float* elnb  = (const float*)d_in[5];
  const float* proto = (const float*)d_in[6];
  const float* ltau  = (const float*)d_in[7];
  const float* flng  = (const float*)d_in[8];
  const float* flnb  = (const float*)d_in[9];
  const int P1 = 10, P2 = 23;
  // per-phase f32 params: wq wk wv wo wo_b w1 b1 w2 b2 lna_g lna_b lnf_g lnf_b
  const float* compat = (const float*)d_in[P2 + 13];
  const float* gammaA = (const float*)d_in[P2 + 14];

  // workspace carve
  size_t off = 0;
  auto alloc = [&](size_t bytes) {
    void* p = (char*)d_ws + off;
    off += (bytes + 255) & ~(size_t)255;
    return p;
  };
  const size_t wDD = 3ull * Dc * Dc, wDF = 3ull * Dc * FFc;
  _Float16* wqh[2]; _Float16* wkh[2]; _Float16* wvh[2]; _Float16* woh[2];
  _Float16* w1h[2]; _Float16* w2h[2];
  for (int ph = 0; ph < 2; ++ph) {
    wqh[ph] = (_Float16*)alloc(wDD * 2); wkh[ph] = (_Float16*)alloc(wDD * 2);
    wvh[ph] = (_Float16*)alloc(wDD * 2); woh[ph] = (_Float16*)alloc(wDD * 2);
    w1h[ph] = (_Float16*)alloc(wDF * 2); w2h[ph] = (_Float16*)alloc(wDF * 2);
  }
  float*    h   = (float*)alloc((size_t)Mc * Dc * 4);
  _Float16* hn  = (_Float16*)alloc((size_t)Mc * Dc * 2);
  _Float16* qb  = (_Float16*)alloc((size_t)Mc * Dc * 2);
  _Float16* kb  = (_Float16*)alloc((size_t)Mc * Dc * 2);
  _Float16* vT  = (_Float16*)alloc((size_t)Mc * Dc * 2);  // [B][Dc][Tc]
  _Float16* ob  = (_Float16*)alloc((size_t)Mc * Dc * 2);
  _Float16* ffb = (_Float16*)alloc((size_t)Mc * FFc * 2);
  float*    Sws = (float*)alloc((size_t)BHc * Tc * Tc * 4);
  _Float16* aP  = (_Float16*)alloc((size_t)BHc * Tc * Tc * 2);
  float*    Ln  = (float*)alloc((size_t)K16 * Dc * 4);
  float*    pcb = (float*)alloc((size_t)BHc * Tc * K16 * 4);

  // d_out layout: H_final | p | s | H_base  (all f32)
  float* Hfinal = (float*)d_out;
  float* Pout   = Hfinal + (size_t)Mc * Dc;
  float* Sout   = Pout + (size_t)Mc * K16;
  float* Hbase  = Sout + (size_t)Mc;

  // transpose-convert all weights to f16 W^T[N][K]
  auto convT = [&](const void* src, _Float16* dst, int Kd, int Nd) {
    f32_to_f16_T_kernel<<<2048, 256, 0, stream>>>((const float*)src, dst, Kd, Nd, 3);
  };
  for (int ph = 0; ph < 2; ++ph) {
    int base = ph ? P2 : P1;
    convT(d_in[base + 0], wqh[ph], Dc, Dc);
    convT(d_in[base + 1], wkh[ph], Dc, Dc);
    convT(d_in[base + 2], wvh[ph], Dc, Dc);
    convT(d_in[base + 3], woh[ph], Dc, Dc);
    convT(d_in[base + 5], w1h[ph], Dc, FFc);   // w1: (D,FF) -> [FF][D]
    convT(d_in[base + 7], w2h[ph], FFc, Dc);   // w2: (FF,D) -> [D][FF]
  }

  // embeddings + LN
  embed_ln_kernel<<<Mc / 8, 256, 0, stream>>>(ids, tok, pos, elng, elnb, h, Mc);

  const dim3 gD(Dc / 64, Mc / 128, 1);
  auto run_layer = [&](int ph, int li) {
    int base = ph ? P2 : P1;
    const float* wo_b  = (const float*)d_in[base + 4];
    const float* b1    = (const float*)d_in[base + 6];
    const float* b2    = (const float*)d_in[base + 8];
    const float* lna_g = (const float*)d_in[base + 9];
    const float* lna_b = (const float*)d_in[base + 10];
    const float* lnf_g = (const float*)d_in[base + 11];
    const float* lnf_b = (const float*)d_in[base + 12];
    size_t oDD = (size_t)li * Dc * Dc, oDF = (size_t)li * Dc * FFc;

    ln_rows_kernel<<<Mc / 8, 256, 0, stream>>>(h, lna_g + li * Dc, lna_b + li * Dc, hn, nullptr, Mc);
    // q,k: [B,T,D] f16 ; v: stored transposed [B][Dc][Tc]
    gemm_f16_kernel<<<gD, 128, 0, stream>>>(hn, wqh[ph] + oDD, nullptr, nullptr, qb, nullptr,
        Mc, Dc, Dc, Dc, Dc, Dc, 0, 0, 0, NHc, 0, 0, 0);
    gemm_f16_kernel<<<gD, 128, 0, stream>>>(hn, wkh[ph] + oDD, nullptr, nullptr, kb, nullptr,
        Mc, Dc, Dc, Dc, Dc, Dc, 0, 0, 0, NHc, 0, 0, 0);
    gemm_f16_kernel<<<gD, 128, 0, stream>>>(hn, wvh[ph] + oDD, nullptr, nullptr, vT, nullptr,
        Mc, Dc, Dc, Dc, Dc, Dc, 0, 0, 0, NHc, 0, 0, 16);
    if (ph) pc_kernel<<<(BHc * Tc * K16) / 256, 256, 0, stream>>>(Pout, compat, pcb, li);
    attn_scores_kernel<<<dim3(Tc / 64, NHc, Bc), 128, 0, stream>>>(
        qb, kb, Sws, pcb, Pout, Sout, gammaA, li, amask, ph);
    softmax_rows_kernel<<<(BHc * Tc) / 8, 256, 0, stream>>>(Sws, aP, (long)BHc * Tc);
    // o[b,t,h*64+d] = sum_j attn[bh,t,j] * vT[b][h*64+d][j]
    gemm_f16_kernel<<<dim3(1, Tc / 128, BHc), 128, 0, stream>>>(aP, vT, nullptr, nullptr, ob, nullptr,
        Tc, DHc, Tc, Tc, Tc, Dc,
        (long)Tc * Tc, (long)Dc * Tc, (long)Tc * Dc,
        NHc, (long)DHc * Tc, (long)DHc, 8);
    gemm_f16_kernel<<<gD, 128, 0, stream>>>(ob, woh[ph] + oDD, wo_b + li * Dc, h, nullptr, nullptr,
        Mc, Dc, Dc, Dc, Dc, Dc, 0, 0, 0, NHc, 0, 0, 1 | 4);
    ln_rows_kernel<<<Mc / 8, 256, 0, stream>>>(h, lnf_g + li * Dc, lnf_b + li * Dc, hn, nullptr, Mc);
    gemm_f16_kernel<<<dim3(FFc / 64, Mc / 128, 1), 128, 0, stream>>>(hn, w1h[ph] + oDF,
        b1 + (size_t)li * FFc, nullptr, ffb, nullptr,
        Mc, FFc, Dc, Dc, Dc, FFc, 0, 0, 0, NHc, 0, 0, 1 | 2);
    gemm_f16_kernel<<<gD, 128, 0, stream>>>(ffb, w2h[ph] + oDF, b2 + li * Dc, h, nullptr, nullptr,
        Mc, Dc, FFc, FFc, FFc, Dc, 0, 0, 0, NHc, 0, 0, 1 | 4);
  };

  // phase 1
  for (int li = 0; li < 3; ++li) run_layer(0, li);

  // H_base, prototypes p, switch s
  (void)hipMemcpyAsync(Hbase, h, sizeof(float) * (size_t)Mc * Dc,
                       hipMemcpyDeviceToDevice, stream);
  ln_rows_kernel<<<Mc / 8, 256, 0, stream>>>(h, flng, flnb, hn, nullptr, Mc);
  proto_norm_kernel<<<K16, 32, 0, stream>>>(proto, Ln);
  proto_p_kernel<<<Mc / 8, 256, 0, stream>>>(hn, Ln, ltau, Pout, Mc);
  switch_s_kernel<<<Mc / 256, 256, 0, stream>>>(Pout, Sout, Mc);

  // phase 2
  for (int li = 0; li < 3; ++li) run_layer(1, li);

  // H_final
  ln_rows_kernel<<<Mc / 8, 256, 0, stream>>>(h, flng, flnb, nullptr, Hfinal, Mc);
}
